// Connectome2HybridActor_5085241279255
// MI455X (gfx1250) — matrix operand order repaired and checked by reading, not compiled, not run on hardware
//
#include <hip/hip_runtime.h>
#include <hip/hip_bf16.h>

// ---------------------------------------------------------------------------
// Connectome RNN actor for gfx1250 (MI455X), bf16 WMMA with f32 accumulation.
//
// Batch rows are independent -> 16 workgroups, each owns a 16-row batch tile
// and runs all T=128 steps with only workgroup barriers. Weights are repacked
// once per call into WMMA-B-operand bf16 tiles in workspace (L2 resident).
// Heads are algebraically fused: Wc = W_ro @ [W_mu|W_ls|W_cl]  (32x5).
// Round 2: wave id forced uniform (scalar loop control around WMMA) and
// 2-tile register blocking (one A fragment feeds two v_wmma).
// ---------------------------------------------------------------------------

typedef __attribute__((ext_vector_type(16))) __bf16 v16bf;
typedef __attribute__((ext_vector_type(8)))  __bf16 v8bf;
typedef __attribute__((ext_vector_type(8)))  float  v8f;

// dims
constexpr int B_   = 256, T_ = 128, OBS = 64;
constexpr int N_ORN = 768, N_PN = 224, N_LN = 128, N_KC = 1728, N_MBON = 32;
constexpr int FEAT = 2880;           // 768+224+128+1728+32
constexpr int INNER = 4;
// feature offsets inside one state row (concat order matches h2)
constexpr int F_ORN = 0, F_PN = 768, F_LN = 992, F_KC = 1120, F_MBON = 2848;

// packed-weight element offsets (bf16 elements), tiles are [nt][kt][512]
constexpr size_t OFF_IN  = 0;        // 64x768    kC=2  nT=48
constexpr size_t OFF_OTO = 49152;    // 768x768   kC=24 nT=48
constexpr size_t OFF_LTO = 638976;   // 128x768   kC=4  nT=48
constexpr size_t OFF_OTP = 737280;   // 768x224   kC=24 nT=14
constexpr size_t OFF_LTP = 909312;   // 128x224   kC=4  nT=14
constexpr size_t OFF_PTP = 937984;   // 224x224   kC=7  nT=14
constexpr size_t OFF_OTL = 988160;   // 768x128   kC=24 nT=8
constexpr size_t OFF_PTL = 1086464;  // 224x128   kC=7  nT=8
constexpr size_t OFF_LTL = 1115136;  // 128x128   kC=4  nT=8
constexpr size_t OFF_KTK = 1131520;  // 1728x1728 kC=54 nT=108
constexpr size_t OFF_MTK = 4117504;  // 32x1728   kC=1  nT=108
constexpr size_t OFF_PTK = 4172800;  // 224x1728  kC=7  nT=108
constexpr size_t OFF_KTM = 4559872;  // 1728x32   kC=54 nT=2
constexpr size_t OFF_WC  = 4615168;  // 32x16 (heads, cols 5..15 zero)
constexpr size_t PK_ELEMS = 4615680;
constexpr size_t PK_BYTES = PK_ELEMS * 2;

// output layout (flat f32): mu(B,T,2) | log_std(B,T,2) | cast(B,T,1) | h2(B,2880)
constexpr size_t OUT_LS = (size_t)B_ * T_ * 2;        // 65536
constexpr size_t OUT_CL = OUT_LS * 2;                 // 131072
constexpr size_t OUT_H2 = OUT_CL + (size_t)B_ * T_;   // 163840

// bf16 <-> f32 helpers (round-to-nearest-even)
__device__ __forceinline__ __bf16 f2bf(float f) {
  unsigned u = __float_as_uint(f);
  unsigned r = (u + 0x7FFFu + ((u >> 16) & 1u)) >> 16;
  unsigned short h = (unsigned short)r;
  return __builtin_bit_cast(__bf16, h);
}
__device__ __forceinline__ float bf2f(__bf16 b) {
  unsigned short h = __builtin_bit_cast(unsigned short, b);
  return __uint_as_float((unsigned)h << 16);
}

// A fragment (16x32 bf16) from LDS, row-major [16][stride].
// Contiguous-k order (same as packed B): lane<16 -> row=lane, K={k0..k0+15}
//             lane>=16 -> row=lane-16, K={k0+16..k0+31}
__device__ __forceinline__ v16bf ldsA(const __bf16* region, int stride, int k0, int lane) {
  const int row = lane & 15;
  const int kh  = (lane & 16) ? 16 : 0;
  const __bf16* p = region + row * stride + k0 + kh;
  v8bf lo = *(const v8bf*)(p);        // ds_load_b128
  v8bf hi = *(const v8bf*)(p + 8);    // ds_load_b128
  return __builtin_shufflevector(lo, hi, 0,1,2,3,4,5,6,7,8,9,10,11,12,13,14,15);
}

// Single-tile accumulate: acc += A(16xK from LDS) * W[:, nt-tile]
__device__ __forceinline__ void mma(v8f& acc, const __bf16* Areg, int stride, int kChunks,
                                    const __bf16* __restrict__ wbase, int nt, int lane) {
  const __bf16* wp = wbase + ((size_t)nt * kChunks) * 512 + lane * 16;
  for (int kt = 0; kt < kChunks; ++kt) {
    v16bf a = ldsA(Areg, stride, kt * 32, lane);
    v16bf b = *(const v16bf*)wp;      // 2x global_load_b128, L2-resident
    wp += 512;
    __builtin_prefetch(wp, 0, 1);     // global_prefetch_b8 for next tile slice
    acc = __builtin_amdgcn_wmma_f32_16x16x32_bf16(
        false, a, false, b, (short)0, acc, false, false);
  }
}

// Pair-blocked accumulate: one shared A fragment feeds two WMMAs (tiles nt,nt+1).
__device__ __forceinline__ void mma2(v8f& acc0, v8f& acc1, const __bf16* Areg, int stride,
                                     int kChunks, const __bf16* __restrict__ wbase,
                                     int nt, int lane) {
  const __bf16* wp0 = wbase + ((size_t)nt * kChunks) * 512 + lane * 16;
  const __bf16* wp1 = wp0 + (size_t)kChunks * 512;
  for (int kt = 0; kt < kChunks; ++kt) {
    v16bf a  = ldsA(Areg, stride, kt * 32, lane);
    v16bf b0 = *(const v16bf*)wp0;
    v16bf b1 = *(const v16bf*)wp1;
    wp0 += 512; wp1 += 512;
    __builtin_prefetch(wp0, 0, 1);
    acc0 = __builtin_amdgcn_wmma_f32_16x16x32_bf16(
        false, a, false, b0, (short)0, acc0, false, false);
    acc1 = __builtin_amdgcn_wmma_f32_16x16x32_bf16(
        false, a, false, b1, (short)0, acc1, false, false);
  }
}

// D layout: element r of lane L is (row = r + (L>=16 ? 8:0), col = L&15).
__device__ __forceinline__ void store_state(__bf16* dst, int stride, int nt, int lane,
                                            const v8f& acc, const float* __restrict__ bias,
                                            const __bf16* extra, int extraStride) {
  const int col = lane & 15;
  const int rb  = (lane & 16) ? 8 : 0;
  const int n   = nt * 16 + col;
  const float bv = bias[n];
#pragma unroll
  for (int r = 0; r < 8; ++r) {
    const int row = rb + r;
    float v = acc[r] + bv;
    if (extra) v += bf2f(extra[row * extraStride + n]);
    dst[row * stride + n] = f2bf(tanhf(v));
  }
}

__device__ __forceinline__ void store_lin(__bf16* dst, int stride, int nt, int lane,
                                          const v8f& acc, const float* __restrict__ bias) {
  const int col = lane & 15;
  const int rb  = (lane & 16) ? 8 : 0;
  const int n   = nt * 16 + col;
  const float bv = bias[n];
#pragma unroll
  for (int r = 0; r < 8; ++r)
    dst[(rb + r) * stride + n] = f2bf(acc[r] + bv);
}

// ---------------------------------------------------------------------------
// Weight repack: f32 row-major W[K][N] -> bf16 WMMA-B tiles [nt][kt][512].
// Within a tile: element (lane,i): k = kt*32 + (lane>=16?16:0) + i, n = nt*16 + (lane&15)
// ---------------------------------------------------------------------------
__global__ void repack_kernel(const float* __restrict__ W, int K, int N,
                              __bf16* __restrict__ dst, int kChunks, int nTiles) {
  size_t e = (size_t)blockIdx.x * blockDim.x + threadIdx.x;
  size_t total = (size_t)nTiles * kChunks * 512;
  if (e >= total) return;
  int within = (int)(e & 511);
  size_t tile = e >> 9;
  int lane = within >> 4, i = within & 15;
  int nt = (int)(tile / kChunks), kt = (int)(tile % kChunks);
  int k = kt * 32 + ((lane & 16) ? 16 : 0) + i;
  int n = nt * 16 + (lane & 15);
  float v = (k < K && n < N) ? W[(size_t)k * N + n] : 0.0f;
  dst[e] = f2bf(v);
}

// Fused heads: Wc[32][5] = W_ro(32x2880) @ [W_mu|W_ls|W_cl], bc[5] = b_ro@[...] + b_head
__global__ void head_combine_kernel(const float* __restrict__ W_ro, const float* __restrict__ b_ro,
                                    const float* __restrict__ W_mu, const float* __restrict__ b_mu,
                                    const float* __restrict__ W_ls, const float* __restrict__ b_ls,
                                    const float* __restrict__ W_cl, const float* __restrict__ b_cl,
                                    __bf16* __restrict__ WcPack, float* __restrict__ bcOut) {
  __shared__ float Wc[32][5];
  const int tid = threadIdx.x;
  if (tid < 160) {
    int m = tid / 5, j = tid % 5;
    float s = 0.f;
    for (int h = 0; h < FEAT; ++h) {
      float col = (j < 2) ? W_mu[h * 2 + j] : (j < 4) ? W_ls[h * 2 + (j - 2)] : W_cl[h];
      s += W_ro[(size_t)m * FEAT + h] * col;
    }
    Wc[m][j] = s;
  } else if (tid < 165) {
    int j = tid - 160;
    float s = (j < 2) ? b_mu[j] : (j < 4) ? b_ls[j - 2] : b_cl[0];
    for (int h = 0; h < FEAT; ++h) {
      float col = (j < 2) ? W_mu[h * 2 + j] : (j < 4) ? W_ls[h * 2 + (j - 2)] : W_cl[h];
      s += b_ro[h] * col;
    }
    bcOut[j] = s;
  }
  __syncthreads();
  for (int e = tid; e < 512; e += 256) {   // pack Wc as one 32x16 B tile
    int lane = e >> 4, i = e & 15;
    int k = ((lane & 16) ? 16 : 0) + i;
    int n = lane & 15;
    WcPack[e] = f2bf((n < 5) ? Wc[k][n] : 0.0f);
  }
}

// ---------------------------------------------------------------------------
// Persistent recurrence kernel: one workgroup per 16-row batch tile.
// 8 waves split the N-tiles of each stage (pairs of tiles per wave);
// ping-pong bf16 state in LDS. Wave id is SGPR-uniform so all tile loops are
// scalar branches and EXEC stays all-ones around every WMMA.
// ---------------------------------------------------------------------------
__global__ __launch_bounds__(256)
void brain_kernel(const float* __restrict__ obs, const __bf16* __restrict__ pk,
                  const float* __restrict__ b_in,  const float* __restrict__ b_orn,
                  const float* __restrict__ b_pn,  const float* __restrict__ b_ln,
                  const float* __restrict__ b_kc,  const float* __restrict__ b_mbon,
                  const float* __restrict__ bc,    float* __restrict__ out) {
  extern __shared__ char smem[];
  __bf16* S0 = (__bf16*)smem;            // [16][FEAT]
  __bf16* S1 = S0 + 16 * FEAT;           // [16][FEAT]
  __bf16* X  = S1 + 16 * FEAT;           // [16][768]  x_t = obs@W_in + b_in
  __bf16* OB = X + 16 * N_ORN;           // [16][64]   staged obs tile

  const int tid  = threadIdx.x;
  const int lane = tid & 31;
  // force SGPR-uniform wave id -> scalar loop control, EXEC all-ones at WMMAs
  const int wave = __builtin_amdgcn_readfirstlane(tid >> 5);
  const int b0   = blockIdx.x * 16;

  for (int i = tid; i < 16 * FEAT; i += 256) S0[i] = f2bf(0.0f);
  __syncthreads();

  __bf16* cur = S0;
  __bf16* nxt = S1;

  for (int t = 0; t < T_; ++t) {
    // stage obs tile as bf16
    for (int i = tid; i < 16 * OBS; i += 256) {
      int r = i >> 6, k = i & 63;
      OB[i] = f2bf(obs[((size_t)(b0 + r) * T_ + t) * OBS + k]);
    }
    __syncthreads();
    // x_t = obs @ W_in + b_in   (48 N-tiles, K=64)
    for (int nt = wave * 2; nt < 48; nt += 16) {
      v8f a0 = {}, a1 = {};
      mma2(a0, a1, OB, OBS, 2, pk + OFF_IN, nt, lane);
      store_lin(X, N_ORN, nt,     lane, a0, b_in);
      store_lin(X, N_ORN, nt + 1, lane, a1, b_in);
    }
    __syncthreads();

    for (int it = 0; it < INNER; ++it) {
      // ORN = tanh(h_orn@W_oto + h_ln@W_lto + x_t + b_orn)   48 tiles
      for (int nt = wave * 2; nt < 48; nt += 16) {
        v8f a0 = {}, a1 = {};
        mma2(a0, a1, cur + F_ORN, FEAT, 24, pk + OFF_OTO, nt, lane);
        mma2(a0, a1, cur + F_LN,  FEAT,  4, pk + OFF_LTO, nt, lane);
        store_state(nxt + F_ORN, FEAT, nt,     lane, a0, b_orn, X, N_ORN);
        store_state(nxt + F_ORN, FEAT, nt + 1, lane, a1, b_orn, X, N_ORN);
      }
      __syncthreads();
      // PN = tanh(orn@W_otp + h_ln@W_ltp + h_pn@W_ptp + b_pn)   14 tiles
      for (int nt = wave * 2; nt < 14; nt += 16) {
        v8f a0 = {}, a1 = {};
        mma2(a0, a1, nxt + F_ORN, FEAT, 24, pk + OFF_OTP, nt, lane);
        mma2(a0, a1, cur + F_LN,  FEAT,  4, pk + OFF_LTP, nt, lane);
        mma2(a0, a1, cur + F_PN,  FEAT,  7, pk + OFF_PTP, nt, lane);
        store_state(nxt + F_PN, FEAT, nt,     lane, a0, b_pn, nullptr, 0);
        store_state(nxt + F_PN, FEAT, nt + 1, lane, a1, b_pn, nullptr, 0);
      }
      __syncthreads();
      // LN = tanh(orn@W_otl + pn@W_ptl + h_ln@W_ltl + b_ln)   8 tiles
      for (int nt = wave * 2; nt < 8; nt += 16) {
        v8f a0 = {}, a1 = {};
        mma2(a0, a1, nxt + F_ORN, FEAT, 24, pk + OFF_OTL, nt, lane);
        mma2(a0, a1, nxt + F_PN,  FEAT,  7, pk + OFF_PTL, nt, lane);
        mma2(a0, a1, cur + F_LN,  FEAT,  4, pk + OFF_LTL, nt, lane);
        store_state(nxt + F_LN, FEAT, nt,     lane, a0, b_ln, nullptr, 0);
        store_state(nxt + F_LN, FEAT, nt + 1, lane, a1, b_ln, nullptr, 0);
      }
      __syncthreads();
      // KC = tanh(h_kc@W_ktk + h_mbon@W_mtk + pn@W_ptk + b_kc)   108 tiles, dominant
      for (int nt = wave * 2; nt < 108; nt += 16) {
        v8f a0 = {}, a1 = {};
        mma2(a0, a1, cur + F_KC,   FEAT, 54, pk + OFF_KTK, nt, lane);
        mma2(a0, a1, cur + F_MBON, FEAT,  1, pk + OFF_MTK, nt, lane);
        mma2(a0, a1, nxt + F_PN,   FEAT,  7, pk + OFF_PTK, nt, lane);
        store_state(nxt + F_KC, FEAT, nt,     lane, a0, b_kc, nullptr, 0);
        store_state(nxt + F_KC, FEAT, nt + 1, lane, a1, b_kc, nullptr, 0);
      }
      __syncthreads();
      // MBON = tanh(kc@W_ktm + b_mbon)   2 tiles (wave 0)
      for (int nt = wave * 2; nt < 2; nt += 16) {
        v8f a0 = {}, a1 = {};
        mma2(a0, a1, nxt + F_KC, FEAT, 54, pk + OFF_KTM, nt, lane);
        store_state(nxt + F_MBON, FEAT, nt,     lane, a0, b_mbon, nullptr, 0);
        store_state(nxt + F_MBON, FEAT, nt + 1, lane, a1, b_mbon, nullptr, 0);
      }
      __syncthreads();
      __bf16* tmp = cur; cur = nxt; nxt = tmp;
    }

    // fused heads: (mbon@W_ro + b_ro)@[W_mu|W_ls|W_cl]+b = mbon@Wc + bc
    if (wave == 0) {   // SGPR-uniform branch: EXEC stays all-ones inside the wave
      v8f acc = {};
      mma(acc, cur + F_MBON, FEAT, 1, pk + OFF_WC, 0, lane);
      const int col = lane & 15;
      const int rb  = (lane & 16) ? 8 : 0;
      if (col < 5) {
        const float bcv = bc[col];
#pragma unroll
        for (int r = 0; r < 8; ++r) {
          const int b = b0 + rb + r;
          const size_t bt = (size_t)b * T_ + t;
          float v = acc[r] + bcv;
          if (col < 2)      out[bt * 2 + col] = v;
          else if (col < 4) out[OUT_LS + bt * 2 + (col - 2)] = fminf(fmaxf(v, -5.0f), 2.0f);
          else              out[OUT_CL + bt] = v;
        }
      }
    }
    __syncthreads();
  }

  // h2 = concat(final states) -> f32
  for (int i = tid; i < 16 * FEAT; i += 256) {
    int r = i / FEAT, f = i % FEAT;
    out[OUT_H2 + (size_t)(b0 + r) * FEAT + f] = bf2f(cur[i]);
  }
}

// ---------------------------------------------------------------------------
extern "C" void kernel_launch(void* const* d_in, const int* in_sizes, int n_in,
                              void* d_out, int out_size, void* d_ws, size_t ws_size,
                              hipStream_t stream) {
  (void)in_sizes; (void)n_in; (void)out_size;
  const size_t need = PK_BYTES + 64;
  if (ws_size < need) return;  // defensive; deterministic for fixed ws_size

  __bf16* pk = (__bf16*)d_ws;
  float*  bc = (float*)((char*)d_ws + PK_BYTES);

  struct RP { int idx; int K; int N; size_t off; };
  const RP rp[13] = {
    {1,   64,  768, OFF_IN },
    {3,  768,  768, OFF_OTO},
    {4,  128,  768, OFF_LTO},
    {5,  768,  224, OFF_OTP},
    {6,  128,  224, OFF_LTP},
    {7,  224,  224, OFF_PTP},
    {8,  768,  128, OFF_OTL},
    {9,  224,  128, OFF_PTL},
    {10, 128,  128, OFF_LTL},
    {11,1728, 1728, OFF_KTK},
    {12,  32, 1728, OFF_MTK},
    {13, 224, 1728, OFF_PTK},
    {14,1728,   32, OFF_KTM},
  };
  for (int i = 0; i < 13; ++i) {
    const int kC = (rp[i].K + 31) / 32, nT = (rp[i].N + 15) / 16;
    const size_t total = (size_t)kC * nT * 512;
    const int blocks = (int)((total + 255) / 256);
    repack_kernel<<<blocks, 256, 0, stream>>>(
        (const float*)d_in[rp[i].idx], rp[i].K, rp[i].N, pk + rp[i].off, kC, nT);
  }
  head_combine_kernel<<<1, 256, 0, stream>>>(
      (const float*)d_in[20], (const float*)d_in[21],
      (const float*)d_in[22], (const float*)d_in[23],
      (const float*)d_in[24], (const float*)d_in[25],
      (const float*)d_in[26], (const float*)d_in[27],
      pk + OFF_WC, bc);

  const size_t smem = (size_t)2 * 16 * FEAT * 2   // S0 + S1
                    + (size_t)16 * N_ORN * 2      // X
                    + (size_t)16 * OBS * 2;       // OB  (~211 KB of 320 KB WGP LDS)
  (void)hipFuncSetAttribute((const void*)&brain_kernel,
                            hipFuncAttributeMaxDynamicSharedMemorySize, (int)smem);
  brain_kernel<<<B_ / 16, 256, smem, stream>>>(
      (const float*)d_in[0], pk,
      (const float*)d_in[2],  (const float*)d_in[15], (const float*)d_in[16],
      (const float*)d_in[17], (const float*)d_in[18], (const float*)d_in[19],
      bc, (float*)d_out);
}
